// Cos_Attn_13503377179165
// MI455X (gfx1250) — matrix operand order repaired and checked by reading, not compiled, not run on hardware
//
#include <hip/hip_runtime.h>
#include <hip/hip_bf16.h>
#include <math.h>

typedef __bf16 bf16_t;
typedef __attribute__((ext_vector_type(16))) __bf16 v16bf;
typedef __attribute__((ext_vector_type(8)))  __bf16 v8bf;
typedef __attribute__((ext_vector_type(8)))  float  v8f;

#define BATCH 4
#define CHN   256
#define NTOK  4096   // 64*64
#define NTILE (NTOK/16)

// ---------------------------------------------------------------------------
// Kernel 1: x [B][C][N] f32  ->  q [B][N][C] bf16   (token-major for WMMA)
// ---------------------------------------------------------------------------
__global__ void cvt_kernel(const float* __restrict__ x, bf16_t* __restrict__ q) {
    const size_t total = (size_t)BATCH * CHN * NTOK;
    for (size_t i = (size_t)blockIdx.x * blockDim.x + threadIdx.x; i < total;
         i += (size_t)gridDim.x * blockDim.x) {
        size_t b = i / ((size_t)CHN * NTOK);
        size_t r = i % ((size_t)CHN * NTOK);
        size_t c = r / NTOK;
        size_t n = r % NTOK;
        q[(b * NTOK + n) * CHN + c] = (bf16_t)x[i];
    }
}

// ---------------------------------------------------------------------------
// Kernel 2: per-token inverse L2 norm over channels (f32, from original x).
// 1e-6 guard vs nm~256 is 4e-9 relative -> separable reciprocal is exact to
// well below f32 epsilon, so store 1/norm and use two mults in the epilogue.
// ---------------------------------------------------------------------------
__global__ void norm_kernel(const float* __restrict__ x, float* __restrict__ inorm) {
    int t = blockIdx.x * blockDim.x + threadIdx.x;       // 0 .. B*N-1
    if (t >= BATCH * NTOK) return;
    int b = t / NTOK, n = t % NTOK;
    const float* p = x + (size_t)b * CHN * NTOK + n;     // stride N over channels
    float s = 0.f;
#pragma unroll 8
    for (int c = 0; c < CHN; ++c) {
        float v = p[(size_t)c * NTOK];
        s += v * v;
    }
    inorm[t] = 1.0f / sqrtf(s);
}

// ---------------------------------------------------------------------------
// Kernel 3: one workgroup per 16-row strip.
//   bf16 WMMA GEMM (K=256), two column tiles in flight (2 independent
//   accumulator chains) -> normalized energy in 256KB LDS -> softmax ->
//   coalesced f32 store.
// ---------------------------------------------------------------------------
__launch_bounds__(256, 1)
__global__ void attn_kernel(const bf16_t* __restrict__ q,
                            const float*  __restrict__ inorm,
                            float*        __restrict__ out) {
    extern __shared__ float smem[];         // 16 * 4096 f32 = 256 KB (dynamic LDS)
    __shared__ float rn_s[16];

    const int b    = blockIdx.x / NTILE;
    const int row0 = (blockIdx.x % NTILE) * 16;
    const int tid  = threadIdx.x;
    const int lane = tid & 31;
    const int wave = tid >> 5;              // 8 waves
    const int lmod = lane & 15;
    const int lhi  = lane >> 4;
    const size_t bN = (size_t)b * NTOK;

    if (tid < 16) rn_s[tid] = inorm[bN + row0 + tid];
    __syncthreads();

    // per-lane inverse row norms matching C/D layout: VGPR v -> M = v + 8*lhi
    float rnl[8];
#pragma unroll
    for (int v = 0; v < 8; ++v) rnl[v] = rn_s[v + 8 * lhi];

    const bf16_t* qb = q + bN * CHN;
    // A fragment base: lane holds row M = row0 + lmod; half-select lhi picks
    // K chunk {lhi*16 .. +15} (contiguous, matching B).
    const bf16_t* pa = qb + (size_t)(row0 + lmod) * CHN + lhi * 16;

    // two column tiles per iteration: wave w owns tiles {2w, 2w+1} + k*16
    for (int t0 = wave * 2; t0 < NTILE; t0 += 16) {
        const int col0 = t0 * 16;
        const int col1 = col0 + 16;
        // B fragment: lane holds column token, K chunk lhi*16..+15
        const bf16_t* pb0 = qb + (size_t)(col0 + lmod) * CHN + lhi * 16;
        const bf16_t* pb1 = qb + (size_t)(col1 + lmod) * CHN + lhi * 16;
        const float cn0 = inorm[bN + col0 + lmod];
        const float cn1 = inorm[bN + col1 + lmod];

        v8f acc0 = {};
        v8f acc1 = {};
#pragma unroll
        for (int kt = 0; kt < 8; ++kt) {
            const int kb = kt * 32;
            union { v16bf v; struct { v8bf lo, hi; } p; } A;
            A.p.lo = *(const v8bf*)(pa + kb);        // K = kb + lhi*16 .. +7
            A.p.hi = *(const v8bf*)(pa + kb + 8);    // K = kb + lhi*16 + 8 .. +15
            v16bf B0 = *(const v16bf*)(pb0 + kb);    // K = kb + lhi*16 .. +15
            v16bf B1 = *(const v16bf*)(pb1 + kb);
            acc0 = __builtin_amdgcn_wmma_f32_16x16x32_bf16(
                false, A.v, false, B0, (short)0, acc0, false, false);
            acc1 = __builtin_amdgcn_wmma_f32_16x16x32_bf16(
                false, A.v, false, B1, (short)0, acc1, false, false);
        }

        // epilogue: scale by invnorm_m*invnorm_n (2 mults), stash strip in LDS
#pragma unroll
        for (int v = 0; v < 8; ++v) {
            const int m = v + 8 * lhi;
            smem[(size_t)m * NTOK + col0 + lmod] = acc0[v] * (rnl[v] * cn0);
            smem[(size_t)m * NTOK + col1 + lmod] = acc1[v] * (rnl[v] * cn1);
        }
    }
    __syncthreads();

    // --- softmax: 16 threads per row, half-wave shuffle reductions ---
    const int r = tid >> 4, sub = tid & 15;
    float* row = smem + (size_t)r * NTOK;

    float mx = -3.402823466e+38f;
    for (int c = sub; c < NTOK; c += 16) mx = fmaxf(mx, row[c]);
#pragma unroll
    for (int o = 8; o >= 1; o >>= 1) mx = fmaxf(mx, __shfl_xor(mx, o, 32));

    float sum = 0.f;
    for (int c = sub; c < NTOK; c += 16) {
        float e = __expf(row[c] - mx);
        row[c] = e;
        sum += e;
    }
#pragma unroll
    for (int o = 8; o >= 1; o >>= 1) sum += __shfl_xor(sum, o, 32);
    const float inv = 1.f / sum;

    float* orow = out + (bN + row0 + r) * (size_t)NTOK;
    for (int c = sub; c < NTOK; c += 16) orow[c] = row[c] * inv;
}

// ---------------------------------------------------------------------------
extern "C" void kernel_launch(void* const* d_in, const int* in_sizes, int n_in,
                              void* d_out, int out_size, void* d_ws, size_t ws_size,
                              hipStream_t stream) {
    const float* x = (const float*)d_in[0];
    float* out = (float*)d_out;

    // workspace: [ q bf16 : B*N*C ][ inv-norms f32 : B*N ]
    bf16_t* q = (bf16_t*)d_ws;
    float* inorm = (float*)((char*)d_ws + (size_t)BATCH * NTOK * CHN * sizeof(bf16_t));

    cvt_kernel<<<4096, 256, 0, stream>>>(x, q);
    norm_kernel<<<(BATCH * NTOK + 255) / 256, 256, 0, stream>>>(x, inorm);
    attn_kernel<<<BATCH * NTILE, 256, (size_t)16 * NTOK * sizeof(float), stream>>>(
        q, inorm, out);
}
